// MambaNeck_38482906972660
// MI455X (gfx1250) — compile-verified, matrix operand order checked by reading, not run on hardware
//
#include <hip/hip_runtime.h>
#include <hip/hip_bf16.h>
#include <math.h>

// ---------------------------------------------------------------------------
// MambaNeck forward for MI455X (gfx1250, wave32, WMMA).
// All GEMMs: bf16 inputs (converted in LDS) with fp32 accumulate via
// v_wmma_f32_16x16x32_bf16.  Everything numerically sensitive (norms, softmax,
// softplus/exp SSM recurrence, SiLU/GELU) stays fp32.
// Workspace requirement: ~290 MB (dominated by the 167.8 MB SSM state h,
// which fits in the 192 MB L2 and stays hot across layers).
// ---------------------------------------------------------------------------

typedef __attribute__((ext_vector_type(16))) __bf16 v16bf;
typedef __attribute__((ext_vector_type(8)))  float  v8f;

#define LDT 40  // LDS row stride (bf16 elems): 80B rows -> 16B-aligned frags

__device__ __forceinline__ unsigned short bf16bits(float f) {
  union { float f; unsigned int u; } in; in.f = f;
  unsigned int r = in.u + 0x7FFFu + ((in.u >> 16) & 1u);  // RNE
  return (unsigned short)(r >> 16);
}

__device__ __forceinline__ __bf16 f2bf(float f) {
  union { unsigned short u; __bf16 b; } out;
  out.u = bf16bits(f);
  return out.b;
}

__device__ __forceinline__ float dev_silu(float x) { return x / (1.0f + expf(-x)); }

// ---------------------------------------------------------------------------
// Generic batched GEMM:  C = epilogue( alpha * (A @ B) )
//   A: M x K, row-major, leading dim lda, batch offset zo*sAo + zi*sAi
//   B: if b_kn==0: stored N x K (weight W, computes A @ W^T)
//      if b_kn==1: stored K x N (computes A @ B directly)
//   epilogue: v = alpha*acc; v += bias[n]; act(softplus); v *= gamma[n];
//             v += res[same layout as C]; C = v
// Block: 128 threads (4 waves), 64x64 tile, BK=32.
// Each wave owns a 32x32 patch = 2x2 WMMA 16x16 tiles.
// All shapes in this model are multiples of 64/32 -> fast paths always taken.
// ---------------------------------------------------------------------------
__global__ __launch_bounds__(128) void gemm_wmma_kernel(
    const float* __restrict__ A, int lda, long sAo, long sAi,
    const float* __restrict__ Bm, int ldb, long sBo, long sBi, int b_kn,
    float* __restrict__ C, int ldc, long sCo, long sCi,
    const float* __restrict__ res,
    const float* __restrict__ bias,
    const float* __restrict__ gamma,
    float alpha, int act, int innerB,
    int M, int N, int K)
{
  __shared__ __bf16 lsA[64 * LDT];
  __shared__ __bf16 lsB[64 * LDT];

  int z  = blockIdx.z;
  int zo = z / innerB, zi = z % innerB;
  A  += (long)zo * sAo + (long)zi * sAi;
  Bm += (long)zo * sBo + (long)zi * sBi;
  long coff = (long)zo * sCo + (long)zi * sCi;
  C += coff;
  if (res) res += coff;

  int m0 = blockIdx.y * 64;
  int n0 = blockIdx.x * 64;

  int t    = threadIdx.x;
  int wave = t >> 5;
  int lane = t & 31;
  int wr = wave >> 1, wc = wave & 1;

  // CDNA5 16-bit fragment K pattern (ISA 7.12.2): lanes 0-15 hold K 0-7,16-23;
  // lanes 16-31 hold K 8-15,24-31.  A indexed by row M, B by col N.
  int half = lane >> 4;
  int r16  = lane & 15;
  int ko1  = half ? 8  : 0;
  int ko2  = half ? 24 : 16;

  bool mfull = (m0 + 64 <= M);
  bool nfull = (n0 + 64 <= N);

  v8f acc[2][2];
  #pragma unroll
  for (int i = 0; i < 2; i++)
    #pragma unroll
    for (int j = 0; j < 2; j++)
      #pragma unroll
      for (int e = 0; e < 8; e++) acc[i][j][e] = 0.0f;

  for (int k0 = 0; k0 < K; k0 += 32) {
    bool kfull = (k0 + 32 <= K);

    // ---- A tile 64x32 -> bf16 LDS ----
    if (mfull && kfull) {
      // vectorized: global_load_b128 + packed ds_store_b64
      #pragma unroll
      for (int q = t; q < 512; q += 128) {
        int row = q >> 3;
        int col = (q & 7) * 4;
        float4 v = *(const float4*)(A + (long)(m0 + row) * lda + k0 + col);
        uint2 pv;
        pv.x = (unsigned int)bf16bits(v.x) | ((unsigned int)bf16bits(v.y) << 16);
        pv.y = (unsigned int)bf16bits(v.z) | ((unsigned int)bf16bits(v.w) << 16);
        *(uint2*)(&lsA[row * LDT + col]) = pv;
      }
    } else {
      for (int idx = t; idx < 64 * 32; idx += 128) {
        int row = idx >> 5, col = idx & 31;
        int gm = m0 + row, gk = k0 + col;
        // branch-free: clamped address + select-zero
        long o = (long)(gm < M ? gm : M - 1) * lda + (gk < K ? gk : K - 1);
        float v = A[o];
        if (!(gm < M && gk < K)) v = 0.0f;
        lsA[row * LDT + col] = f2bf(v);
      }
    }

    // ---- B tile -> lsB[n][k] ----
    if (b_kn == 0) {
      if (nfull && kfull) {
        #pragma unroll
        for (int q = t; q < 512; q += 128) {
          int n   = q >> 3;
          int col = (q & 7) * 4;
          float4 v = *(const float4*)(Bm + (long)(n0 + n) * ldb + k0 + col);
          uint2 pv;
          pv.x = (unsigned int)bf16bits(v.x) | ((unsigned int)bf16bits(v.y) << 16);
          pv.y = (unsigned int)bf16bits(v.z) | ((unsigned int)bf16bits(v.w) << 16);
          *(uint2*)(&lsB[n * LDT + col]) = pv;
        }
      } else {
        for (int idx = t; idx < 64 * 32; idx += 128) {
          int n = idx >> 5, kk = idx & 31;
          int gn = n0 + n, gk = k0 + kk;
          long o = (long)(gn < N ? gn : N - 1) * ldb + (gk < K ? gk : K - 1);
          float v = Bm[o];
          if (!(gn < N && gk < K)) v = 0.0f;
          lsB[n * LDT + kk] = f2bf(v);
        }
      }
    } else {
      if (nfull && kfull) {
        #pragma unroll
        for (int q = t; q < 512; q += 128) {
          int kk = q >> 4;
          int n4 = (q & 15) * 4;
          float4 v = *(const float4*)(Bm + (long)(k0 + kk) * ldb + n0 + n4);
          lsB[(n4 + 0) * LDT + kk] = f2bf(v.x);
          lsB[(n4 + 1) * LDT + kk] = f2bf(v.y);
          lsB[(n4 + 2) * LDT + kk] = f2bf(v.z);
          lsB[(n4 + 3) * LDT + kk] = f2bf(v.w);
        }
      } else {
        for (int idx = t; idx < 64 * 32; idx += 128) {
          int n = idx & 63, kk = idx >> 6;
          int gn = n0 + n, gk = k0 + kk;
          long o = (long)(gk < K ? gk : K - 1) * ldb + (gn < N ? gn : N - 1);
          float v = Bm[o];
          if (!(gn < N && gk < K)) v = 0.0f;
          lsB[n * LDT + kk] = f2bf(v);
        }
      }
    }

    // Prefetch next A K-tile into cache (global_prefetch_b8)
    if (k0 + 32 < K) {
      int pr = m0 + (t & 63);
      if (pr < M) __builtin_prefetch(&A[(long)pr * lda + k0 + 32], 0, 1);
    }
    __syncthreads();

    #pragma unroll
    for (int i = 0; i < 2; i++) {
      int tm = wr * 32 + i * 16;
      v16bf a;
      #pragma unroll
      for (int e = 0; e < 8; e++) {
        a[e]     = lsA[(tm + r16) * LDT + ko1 + e];
        a[e + 8] = lsA[(tm + r16) * LDT + ko2 + e];
      }
      #pragma unroll
      for (int j = 0; j < 2; j++) {
        int tn = wc * 32 + j * 16;
        v16bf b;
        #pragma unroll
        for (int e = 0; e < 8; e++) {
          b[e]     = lsB[(tn + r16) * LDT + ko1 + e];
          b[e + 8] = lsB[(tn + r16) * LDT + ko2 + e];
        }
        acc[i][j] = __builtin_amdgcn_wmma_f32_16x16x32_bf16(
            false, a, false, b, (short)0, acc[i][j], false, false);
      }
    }
    __syncthreads();
  }

  // Epilogue.  C frag layout: lane n = lane&15; VGPR r -> M = r + 8*(lane>>4).
  #pragma unroll
  for (int i = 0; i < 2; i++)
    #pragma unroll
    for (int j = 0; j < 2; j++) {
      int tm = wr * 32 + i * 16, tn = wc * 32 + j * 16;
      int gn = n0 + tn + r16;
      if (mfull && nfull) {
        #pragma unroll
        for (int rr = 0; rr < 8; rr++) {
          int gm = m0 + tm + half * 8 + rr;
          float v = alpha * acc[i][j][rr];
          if (bias)  v += bias[gn];
          if (act == 1) v = fmaxf(v, 0.0f) + log1pf(expf(-fabsf(v)));  // softplus
          if (gamma) v *= gamma[gn];
          long o = (long)gm * ldc + gn;
          if (res) v += res[o];
          C[o] = v;
        }
      } else {
        #pragma unroll
        for (int rr = 0; rr < 8; rr++) {
          int gm = m0 + tm + half * 8 + rr;
          if (gm < M && gn < N) {
            float v = alpha * acc[i][j][rr];
            if (bias)  v += bias[gn];
            if (act == 1) v = fmaxf(v, 0.0f) + log1pf(expf(-fabsf(v)));
            if (gamma) v *= gamma[gn];
            long o = (long)gm * ldc + gn;
            if (res) v += res[o];
            C[o] = v;
          }
        }
      }
    }
}

// ---------------------------------------------------------------------------
__global__ __launch_bounds__(256) void rmsnorm_kernel(
    const float* __restrict__ x, const float* __restrict__ w,
    float* __restrict__ o, int D, float eps)
{
  __shared__ float red[256];
  const float* xr = x + (long)blockIdx.x * D;
  float* orow = o + (long)blockIdx.x * D;
  int t = threadIdx.x;
  float s = 0.0f;
  for (int i = t; i < D; i += 256) { float v = xr[i]; s += v * v; }
  red[t] = s; __syncthreads();
  for (int st = 128; st; st >>= 1) { if (t < st) red[t] += red[t + st]; __syncthreads(); }
  float scale = rsqrtf(red[0] / (float)D + eps);
  for (int i = t; i < D; i += 256) orow[i] = xr[i] * scale * w[i];
}

__global__ __launch_bounds__(256) void layernorm_kernel(
    const float* __restrict__ x, const float* __restrict__ w,
    const float* __restrict__ b, float* __restrict__ o, int D, float eps)
{
  __shared__ float r1[256], r2[256];
  const float* xr = x + (long)blockIdx.x * D;
  float* orow = o + (long)blockIdx.x * D;
  int t = threadIdx.x;
  float s = 0.0f, s2 = 0.0f;
  for (int i = t; i < D; i += 256) { float v = xr[i]; s += v; s2 += v * v; }
  r1[t] = s; r2[t] = s2; __syncthreads();
  for (int st = 128; st; st >>= 1) {
    if (t < st) { r1[t] += r1[t + st]; r2[t] += r2[t + st]; }
    __syncthreads();
  }
  float mu  = r1[0] / (float)D;
  float var = r2[0] / (float)D - mu * mu;
  float inv = rsqrtf(var + eps);
  for (int i = t; i < D; i += 256) orow[i] = (xr[i] - mu) * inv * w[i] + b[i];
}

// depthwise conv1d (k=3, SAME) over L on first half of xz channels, + SiLU
__global__ __launch_bounds__(256) void conv1d_silu_kernel(
    const float* __restrict__ xz, const float* __restrict__ w,
    const float* __restrict__ b, float* __restrict__ x1s, int Bsz, int L)
{
  long id = (long)blockIdx.x * 256 + threadIdx.x;
  long total = (long)Bsz * L * 1024;
  if (id >= total) return;
  int c  = (int)(id & 1023);
  long bl = id >> 10;
  int l  = (int)(bl % L);
  int bb = (int)(bl / L);
  float acc = b[c];
  #pragma unroll
  for (int k = 0; k < 3; k++) {
    int ll = l + k - 1;
    if (ll >= 0 && ll < L)
      acc += xz[((long)(bb * L + ll)) * 2048 + c] * w[c * 3 + k];
  }
  x1s[id] = dev_silu(acc);
}

// SSM elementwise update (per token, per channel; 16-wide state loop).
// h layout: (token, e, 16) fp32.  Traffic-dominant tensor (167.8MB, L2-resident)
__global__ __launch_bounds__(256) void mamba_state_kernel(
    const float* __restrict__ delta, const float* __restrict__ x1s,
    const float* __restrict__ xz, const float* __restrict__ xdbl,
    const float* __restrict__ A_log, const float* __restrict__ Dp,
    float* __restrict__ h, float* __restrict__ yg, long total)
{
  long id = (long)blockIdx.x * 256 + threadIdx.x;
  if (id >= total) return;
  int e   = (int)(id & 1023);
  long tk = id >> 10;
  float d  = delta[id];
  float x1 = x1s[id];
  float z  = xz[tk * 2048 + 1024 + e];
  const float* Bp = xdbl + tk * 64 + 32;
  const float* Cp = xdbl + tk * 64 + 48;
  float* hp = h + id * 16;
  float y = 0.0f;
  #pragma unroll
  for (int n = 0; n < 16; n++) {
    float Av = -expf(A_log[e * 16 + n]);
    float hn = expf(d * Av) * hp[n] + d * Bp[n] * x1;
    hp[n] = hn;
    y += hn * Cp[n];
  }
  y += Dp[e] * x1;
  yg[id] = y * dev_silu(z);
}

__global__ __launch_bounds__(128) void softmax_kernel(float* __restrict__ s, int len)
{
  __shared__ float red[128];
  float* p = s + (long)blockIdx.x * len;
  int t = threadIdx.x;
  float m = -3.4e38f;
  for (int i = t; i < len; i += 128) m = fmaxf(m, p[i]);
  red[t] = m; __syncthreads();
  for (int st = 64; st; st >>= 1) { if (t < st) red[t] = fmaxf(red[t], red[t + st]); __syncthreads(); }
  m = red[0]; __syncthreads();
  float sum = 0.0f;
  for (int i = t; i < len; i += 128) { float e = expf(p[i] - m); p[i] = e; sum += e; }
  red[t] = sum; __syncthreads();
  for (int st = 64; st; st >>= 1) { if (t < st) red[t] += red[t + st]; __syncthreads(); }
  float inv = 1.0f / red[0];
  for (int i = t; i < len; i += 128) p[i] *= inv;
}

// depthwise 3x3 conv (SAME) on (B,C=128,16,16) images + exact GELU.
// tokens laid out (b, l=i*16+j, c)
__global__ __launch_bounds__(256) void dwconv3x3_gelu_kernel(
    const float* __restrict__ h1, const float* __restrict__ w,
    const float* __restrict__ b, float* __restrict__ h2, int Bsz)
{
  int id = blockIdx.x * 256 + threadIdx.x;
  int total = Bsz * 256 * 128;
  if (id >= total) return;
  int c  = id & 127;
  int l  = (id >> 7) & 255;
  int bb = id >> 15;
  int i = l >> 4, j = l & 15;
  float acc = b[c];
  #pragma unroll
  for (int ki = 0; ki < 3; ki++)
    #pragma unroll
    for (int kj = 0; kj < 3; kj++) {
      int ii = i + ki - 1, jj = j + kj - 1;
      if (ii >= 0 && ii < 16 && jj >= 0 && jj < 16)
        acc += h1[((long)bb * 256 + ii * 16 + jj) * 128 + c] * w[c * 9 + ki * 3 + kj];
    }
  h2[id] = 0.5f * acc * (1.0f + erff(acc * 0.70710678118654752f));
}

// ---------------------------------------------------------------------------
extern "C" void kernel_launch(void* const* d_in, const int* in_sizes, int n_in,
                              void* d_out, int out_size, void* d_ws, size_t ws_size,
                              hipStream_t stream)
{
  (void)in_sizes; (void)n_in; (void)out_size; (void)ws_size;
  const int B = 8, L = 320, LS = 256, D = 512, ED = 1024, NL = 4;
  const int T = B * L, TS = B * LS;  // 2560, 2048 tokens

  // ---- parameter pointers: setup_inputs() dict insertion order ----
  int idx = 0;
  auto nf = [&]() -> const float* { return (const float*)d_in[idx++]; };
  const float* X  = nf();
  const float* XS = nf();
  struct MambaP { const float *in_w,*conv_w,*conv_b,*xp_w,*dt_w,*dt_b,*A_log,*Dd,*out_w; };
  struct AttnP  { const float *in_w,*in_b,*out_w,*out_b; };
  struct InjP   { const float *lnq_w,*lnq_b,*lnf_w,*lnf_b; AttnP a; const float* gamma; };
  struct ExtP   { const float *lnq_w,*lnq_b,*lnf_w,*lnf_b,*lnffn_w,*lnffn_b; AttnP a;
                  const float *fc1_w,*fc1_b,*dw_w,*dw_b,*fc2_w,*fc2_b; };
  const float* rms_w[4]; MambaP mp[4];
  for (int i = 0; i < NL; i++) {
    rms_w[i] = nf();
    mp[i].in_w = nf(); mp[i].conv_w = nf(); mp[i].conv_b = nf(); mp[i].xp_w = nf();
    mp[i].dt_w = nf(); mp[i].dt_b = nf(); mp[i].A_log = nf(); mp[i].Dd = nf(); mp[i].out_w = nf();
  }
  InjP inj[4]; ExtP ext[4][3]; const int n_ext[4] = {1, 1, 1, 3};
  for (int i = 0; i < NL; i++) {
    inj[i].lnq_w = nf(); inj[i].lnq_b = nf(); inj[i].lnf_w = nf(); inj[i].lnf_b = nf();
    inj[i].a.in_w = nf(); inj[i].a.in_b = nf(); inj[i].a.out_w = nf(); inj[i].a.out_b = nf();
    inj[i].gamma = nf();
    for (int e = 0; e < n_ext[i]; e++) {
      ExtP& E = ext[i][e];
      E.lnq_w = nf(); E.lnq_b = nf(); E.lnf_w = nf(); E.lnf_b = nf();
      E.lnffn_w = nf(); E.lnffn_b = nf();
      E.a.in_w = nf(); E.a.in_b = nf(); E.a.out_w = nf(); E.a.out_b = nf();
      E.fc1_w = nf(); E.fc1_b = nf(); E.dw_w = nf(); E.dw_b = nf();
      E.fc2_w = nf(); E.fc2_b = nf();
    }
  }
  const float* norm_f = nf();

  // ---- workspace carve-up (~290 MB) ----
  float* wsp = (float*)d_ws;
  size_t off = 0;
  auto alloc = [&](size_t n) { float* p = wsp + off; off += n; return p; };
  float* h      = alloc((size_t)T * ED * 16);   // 167.8 MB SSM state
  float* cur_x  = alloc((size_t)T * D);
  float* cur_xs = alloc((size_t)TS * D);
  float* xr     = alloc((size_t)T * D);
  float* xz     = alloc((size_t)T * 2 * ED);
  float* x1s    = alloc((size_t)T * ED);
  float* xdbl   = alloc((size_t)T * 64);
  float* delta  = alloc((size_t)T * ED);
  float* yg     = alloc((size_t)T * ED);
  float* lnq    = alloc((size_t)T * D);
  float* lnf    = alloc((size_t)T * D);
  float* qh     = alloc((size_t)T * D);
  float* kh     = alloc((size_t)T * D);
  float* vh     = alloc((size_t)T * D);
  float* sc     = alloc((size_t)64 * L * LS);   // batched attention scores
  float* ah     = alloc((size_t)T * D);
  float* f1     = alloc((size_t)TS * 128);
  float* f2     = alloc((size_t)TS * 128);

  auto gemm = [&](const float* A, int lda, long sAo, long sAi,
                  const float* Bmat, int ldb, long sBo, long sBi, int b_kn,
                  float* C, int ldc, long sCo, long sCi,
                  const float* res, const float* bias, const float* gamma,
                  float alpha, int act, int innerB, int batch,
                  int M, int N, int K) {
    dim3 grid((N + 63) / 64, (M + 63) / 64, batch);
    gemm_wmma_kernel<<<grid, 128, 0, stream>>>(A, lda, sAo, sAi, Bmat, ldb, sBo, sBi, b_kn,
        C, ldc, sCo, sCi, res, bias, gamma, alpha, act, innerB, M, N, K);
  };

  // ---- init ----
  hipMemsetAsync(h, 0, (size_t)T * ED * 16 * sizeof(float), stream);
  hipMemcpyAsync(cur_x,  X,  (size_t)T  * D * sizeof(float), hipMemcpyDeviceToDevice, stream);
  hipMemcpyAsync(cur_xs, XS, (size_t)TS * D * sizeof(float), hipMemcpyDeviceToDevice, stream);

  for (int i = 0; i < NL; i++) {
    // ================= Mamba block =================
    rmsnorm_kernel<<<T, 256, 0, stream>>>(cur_x, rms_w[i], xr, D, 1e-5f);
    // xz = xr @ in_w^T                         (2560 x 2048 x 512)
    gemm(xr, D, 0, 0, mp[i].in_w, D, 0, 0, 0, xz, 2 * ED, 0, 0,
         nullptr, nullptr, nullptr, 1.0f, 0, 1, 1, T, 2 * ED, D);
    conv1d_silu_kernel<<<((long)T * ED + 255) / 256, 256, 0, stream>>>(
        xz, mp[i].conv_w, mp[i].conv_b, x1s, B, L);
    // x_dbl = x1 @ xp_w^T                      (2560 x 64 x 1024)
    gemm(x1s, ED, 0, 0, mp[i].xp_w, ED, 0, 0, 0, xdbl, 64, 0, 0,
         nullptr, nullptr, nullptr, 1.0f, 0, 1, 1, T, 64, ED);
    // delta = softplus(dt @ dt_w^T + dt_b)     (2560 x 1024 x 32), dt strided in x_dbl
    gemm(xdbl, 64, 0, 0, mp[i].dt_w, 32, 0, 0, 0, delta, ED, 0, 0,
         nullptr, mp[i].dt_b, nullptr, 1.0f, 1, 1, 1, T, ED, 32);
    mamba_state_kernel<<<((long)T * ED + 255) / 256, 256, 0, stream>>>(
        delta, x1s, xz, xdbl, mp[i].A_log, mp[i].Dd, h, yg, (long)T * ED);
    // x += yg @ out_w^T                        (2560 x 512 x 1024)
    gemm(yg, ED, 0, 0, mp[i].out_w, ED, 0, 0, 0, cur_x, D, 0, 0,
         cur_x, nullptr, nullptr, 1.0f, 0, 1, 1, T, D, ED);

    // ================= Injector: x attends to xs =================
    {
      const InjP& J = inj[i];
      layernorm_kernel<<<T,  256, 0, stream>>>(cur_x,  J.lnq_w, J.lnq_b, lnq, D, 1e-6f);
      layernorm_kernel<<<TS, 256, 0, stream>>>(cur_xs, J.lnf_w, J.lnf_b, lnf, D, 1e-6f);
      gemm(lnq, D, 0, 0, J.a.in_w,             D, 0, 0, 0, qh, D, 0, 0,
           nullptr, J.a.in_b,        nullptr, 1.0f, 0, 1, 1, T,  D, D);
      gemm(lnf, D, 0, 0, J.a.in_w + 512 * 512, D, 0, 0, 0, kh, D, 0, 0,
           nullptr, J.a.in_b + 512,  nullptr, 1.0f, 0, 1, 1, TS, D, D);
      gemm(lnf, D, 0, 0, J.a.in_w + 2*512*512, D, 0, 0, 0, vh, D, 0, 0,
           nullptr, J.a.in_b + 1024, nullptr, 1.0f, 0, 1, 1, TS, D, D);
      // scores[b,h] = 0.125 * q(320x64) @ k(256x64)^T, batched over 64 (b,h)
      gemm(qh, D, (long)L * D, 64, kh, D, (long)LS * D, 64, 0,
           sc, LS, (long)8 * L * LS, (long)L * LS,
           nullptr, nullptr, nullptr, 0.125f, 0, 8, 64, L, LS, 64);
      softmax_kernel<<<64 * L, 128, 0, stream>>>(sc, LS);
      // out[b,h] = attn(320x256) @ v(256x64)   (B given as KxN)
      gemm(sc, LS, (long)8 * L * LS, (long)L * LS, vh, D, (long)LS * D, 64, 1,
           ah, D, (long)L * D, 64,
           nullptr, nullptr, nullptr, 1.0f, 0, 8, 64, L, 64, LS);
      // x = x + gamma * (ah @ out_w^T + out_b)
      gemm(ah, D, 0, 0, J.a.out_w, D, 0, 0, 0, cur_x, D, 0, 0,
           cur_x, J.a.out_b, J.gamma, 1.0f, 0, 1, 1, T, D, D);
    }

    // ================= Extractors: xs attends to x, then conv-FFN ==========
    for (int e = 0; e < n_ext[i]; e++) {
      const ExtP& E = ext[i][e];
      layernorm_kernel<<<TS, 256, 0, stream>>>(cur_xs, E.lnq_w, E.lnq_b, lnq, D, 1e-6f);
      layernorm_kernel<<<T,  256, 0, stream>>>(cur_x,  E.lnf_w, E.lnf_b, lnf, D, 1e-6f);
      gemm(lnq, D, 0, 0, E.a.in_w,             D, 0, 0, 0, qh, D, 0, 0,
           nullptr, E.a.in_b,        nullptr, 1.0f, 0, 1, 1, TS, D, D);
      gemm(lnf, D, 0, 0, E.a.in_w + 512 * 512, D, 0, 0, 0, kh, D, 0, 0,
           nullptr, E.a.in_b + 512,  nullptr, 1.0f, 0, 1, 1, T,  D, D);
      gemm(lnf, D, 0, 0, E.a.in_w + 2*512*512, D, 0, 0, 0, vh, D, 0, 0,
           nullptr, E.a.in_b + 1024, nullptr, 1.0f, 0, 1, 1, T,  D, D);
      // scores: q 256x64 vs k 320x64 per (b,h)
      gemm(qh, D, (long)LS * D, 64, kh, D, (long)L * D, 64, 0,
           sc, L, (long)8 * LS * L, (long)LS * L,
           nullptr, nullptr, nullptr, 0.125f, 0, 8, 64, LS, L, 64);
      softmax_kernel<<<64 * LS, 128, 0, stream>>>(sc, L);
      gemm(sc, L, (long)8 * LS * L, (long)LS * L, vh, D, (long)L * D, 64, 1,
           ah, D, (long)LS * D, 64,
           nullptr, nullptr, nullptr, 1.0f, 0, 8, 64, LS, 64, L);
      // xs = xs + (ah @ out_w^T + out_b)
      gemm(ah, D, 0, 0, E.a.out_w, D, 0, 0, 0, cur_xs, D, 0, 0,
           cur_xs, E.a.out_b, nullptr, 1.0f, 0, 1, 1, TS, D, D);
      // conv-FFN: LN -> fc1 -> dw3x3+gelu -> fc2 (+residual)
      layernorm_kernel<<<TS, 256, 0, stream>>>(cur_xs, E.lnffn_w, E.lnffn_b, lnq, D, 1e-6f);
      gemm(lnq, D, 0, 0, E.fc1_w, D, 0, 0, 0, f1, 128, 0, 0,
           nullptr, E.fc1_b, nullptr, 1.0f, 0, 1, 1, TS, 128, D);
      dwconv3x3_gelu_kernel<<<(TS * 128 + 255) / 256, 256, 0, stream>>>(
          f1, E.dw_w, E.dw_b, f2, B);
      gemm(f2, 128, 0, 0, E.fc2_w, 128, 0, 0, 0, cur_xs, D, 0, 0,
           cur_xs, E.fc2_b, nullptr, 1.0f, 0, 1, 1, TS, D, 128);
    }
  }

  // ---- outputs: (rms(x, norm_f), xs) concatenated ----
  rmsnorm_kernel<<<T, 256, 0, stream>>>(cur_x, norm_f, (float*)d_out, D, 1e-5f);
  hipMemcpyAsync((float*)d_out + (size_t)T * D, cur_xs,
                 (size_t)TS * D * sizeof(float), hipMemcpyDeviceToDevice, stream);
}
